// ASAP_81243601371620
// MI455X (gfx1250) — matrix operand order repaired and checked by reading, not compiled, not run on hardware
//
#include <hip/hip_runtime.h>

// ---------------------------------------------------------------------------
// ASAP-pool GNN forward for MI455X (gfx1250, wave32).
// All matmuls run through v_wmma_f32_16x16x32_bf16.
// ---------------------------------------------------------------------------

#define N0   2048
#define G    16
#define HID  128
#define LD   2048   // leading dim used for all big square-ish buffers

typedef __attribute__((ext_vector_type(16))) __bf16 v16bf;
typedef __attribute__((ext_vector_type(4)))  __bf16 v4bf;
typedef __attribute__((ext_vector_type(8)))  float  v8f;

#define BM 64
#define BN 64
#define BK 64   // two 16x16x32 K-steps staged per barrier

// ------------------------------ WMMA GEMM ----------------------------------
// C[M,N] = epilogue( op(A)[M,K] @ op(B)[K,N] )
// 256 threads = 8 wave32s: 4 M-tiles x 2 N-wave-groups; each wave owns two
// 16x16 accumulators and issues 4 WMMAs per barrier pair (BK=64).
//
// LDS tiles are stored pre-permuted into the ISA 7.12.2 per-lane fragment
// layout so each lane reads its 16xbf16 fragment as one contiguous 32-byte
// vector (2x ds_load_b128).  Staging issues all global loads first (clause),
// then converts f32->bf16 and scatters into LDS.
//
// A-fragment map for element (r,c) of the MxK tile:
//   khalf = c>>5, lane = ((c>>3)&1)*16 + (r&15),
//   i     = ((c>>4)&1)*8 + ((c>>1)&3)*2 + (c&1)
// B-fragment map for element (r,c) of the KxN tile:
//   khalf = r>>5, lane = ((r>>3)&1)*16 + (c&15), i = ((r>>4)&1)*8 + (r&7)
template <int TA, int TB>
__global__ __launch_bounds__(256) void k_gemm(
    const float* __restrict__ A, int lda,
    const float* __restrict__ B, int ldb,
    const float* __restrict__ Cacc,
    float* __restrict__ Cout, int ldc,
    int M, int N, int K,
    const float* __restrict__ rowdiv,
    const float* __restrict__ bias,
    int relu)
{
    __shared__ alignas(32) __bf16 AsF[2][4][32][16];   // [khalf][m-tile][lane][elem]
    __shared__ alignas(32) __bf16 BsF[2][4][32][16];   // [khalf][n-tile][lane][elem]

    const int tid  = threadIdx.x;
    const int lane = tid & 31;
    const int wave = tid >> 5;
    const int wm   = wave >> 1;          // 0..3 : M tile
    const int wn   = wave & 1;           // 0..1 : N tile pair (2 subtiles)
    const int bm   = blockIdx.y * BM;
    const int bn   = blockIdx.x * BN;

    v8f acc0 = {0.f,0.f,0.f,0.f,0.f,0.f,0.f,0.f};
    v8f acc1 = {0.f,0.f,0.f,0.f,0.f,0.f,0.f,0.f};

    const bool fullA = (bm + BM <= M);
    const bool fullB = (bn + BN <= N);

    for (int k0 = 0; k0 < K; k0 += BK) {
        const bool fullK = (k0 + BK <= K);

        // ---- stage A tile (64x64) into fragment layout -------------------
        if (fullA && fullK) {
            float4 fr[4];
#pragma unroll
            for (int t = 0; t < 4; ++t) {           // issue loads (clause)
                const int vid = tid + t * 256;      // 0..1023 float4 slots
                if (TA == 0) {
                    const int r = vid >> 4, c4 = (vid & 15) * 4;
                    fr[t] = *(const float4*)&A[(size_t)(bm + r) * lda + k0 + c4];
                } else {
                    const int c = vid >> 4, r4 = (vid & 15) * 4;
                    fr[t] = *(const float4*)&A[(size_t)(k0 + c) * lda + bm + r4];
                }
            }
#pragma unroll
            for (int t = 0; t < 4; ++t) {           // convert + LDS store
                const int vid = tid + t * 256;
                if (TA == 0) {                      // contiguous c -> b64 store
                    const int r = vid >> 4, c4 = (vid & 15) * 4;
                    const v4bf pk = {(__bf16)fr[t].x, (__bf16)fr[t].y,
                                     (__bf16)fr[t].z, (__bf16)fr[t].w};
                    const int ib = ((c4 >> 4) & 1) * 8 + ((c4 >> 1) & 3) * 2;
                    *(v4bf*)&AsF[c4 >> 5][r >> 4][((c4 >> 3) & 1) * 16 + (r & 15)][ib] = pk;
                } else {                            // scattered lanes
                    const int c = vid >> 4, r4 = (vid & 15) * 4;
                    const float fv[4] = {fr[t].x, fr[t].y, fr[t].z, fr[t].w};
#pragma unroll
                    for (int j = 0; j < 4; ++j) {
                        const int r = r4 + j;
                        AsF[c >> 5][r >> 4][((c >> 3) & 1) * 16 + (r & 15)]
                           [((c >> 4) & 1) * 8 + ((c >> 1) & 3) * 2 + (c & 1)] = (__bf16)fv[j];
                    }
                }
            }
        } else {
            float fv[16];
#pragma unroll
            for (int t = 0; t < 16; ++t) {          // guarded loads (clause)
                const int idx = tid + t * 256;      // 0..4095
                const int r = idx >> 6, c = idx & 63;
                const int gr = bm + r, gk = k0 + c;
                fv[t] = 0.f;
                if (gr < M && gk < K)
                    fv[t] = (TA == 0) ? A[(size_t)gr * lda + gk] : A[(size_t)gk * lda + gr];
            }
#pragma unroll
            for (int t = 0; t < 16; ++t) {
                const int idx = tid + t * 256;
                const int r = idx >> 6, c = idx & 63;
                AsF[c >> 5][r >> 4][((c >> 3) & 1) * 16 + (r & 15)]
                   [((c >> 4) & 1) * 8 + ((c >> 1) & 3) * 2 + (c & 1)] = (__bf16)fv[t];
            }
        }

        // ---- stage B tile (64x64) into fragment layout -------------------
        if (fullB && fullK) {
            float4 fr[4];
#pragma unroll
            for (int t = 0; t < 4; ++t) {           // issue loads (clause)
                const int vid = tid + t * 256;
                if (TB == 0) {
                    const int r = vid >> 4, c4 = (vid & 15) * 4;
                    fr[t] = *(const float4*)&B[(size_t)(k0 + r) * ldb + bn + c4];
                } else {
                    const int c = vid >> 4, r4 = (vid & 15) * 4;
                    fr[t] = *(const float4*)&B[(size_t)(bn + c) * ldb + k0 + r4];
                }
            }
#pragma unroll
            for (int t = 0; t < 4; ++t) {           // convert + LDS store
                const int vid = tid + t * 256;
                if (TB == 0) {                      // scattered lanes
                    const int r = vid >> 4, c4 = (vid & 15) * 4;
                    const float fv[4] = {fr[t].x, fr[t].y, fr[t].z, fr[t].w};
#pragma unroll
                    for (int j = 0; j < 4; ++j) {
                        const int c = c4 + j;
                        BsF[r >> 5][c >> 4][((r >> 3) & 1) * 16 + (c & 15)][((r >> 4) & 1) * 8 + (r & 7)] = (__bf16)fv[j];
                    }
                } else {                            // contiguous r -> b64 store
                    const int c = vid >> 4, r4 = (vid & 15) * 4;
                    const v4bf pk = {(__bf16)fr[t].x, (__bf16)fr[t].y,
                                     (__bf16)fr[t].z, (__bf16)fr[t].w};
                    *(v4bf*)&BsF[r4 >> 5][c >> 4][((r4 >> 3) & 1) * 16 + (c & 15)][((r4 >> 4) & 1) * 8 + (r4 & 7)] = pk;
                }
            }
        } else {
            float fv[16];
#pragma unroll
            for (int t = 0; t < 16; ++t) {          // guarded loads (clause)
                const int idx = tid + t * 256;
                const int r = idx >> 6, c = idx & 63;
                const int gk = k0 + r, gc = bn + c;
                fv[t] = 0.f;
                if (gk < K && gc < N)
                    fv[t] = (TB == 0) ? B[(size_t)gk * ldb + gc] : B[(size_t)gc * ldb + gk];
            }
#pragma unroll
            for (int t = 0; t < 16; ++t) {
                const int idx = tid + t * 256;
                const int r = idx >> 6, c = idx & 63;
                BsF[r >> 5][c >> 4][((r >> 3) & 1) * 16 + (c & 15)][((r >> 4) & 1) * 8 + (r & 7)] = (__bf16)fv[t];
            }
        }

        // hint next K tile toward the caches (global_prefetch_b8)
        const int kn = k0 + BK;
        if (kn < K && fullA) {
            const int gr = bm + (tid & 63);
            __builtin_prefetch((TA == 0) ? &A[(size_t)gr * lda + kn]
                                         : &A[(size_t)kn * lda + gr], 0, 1);
        }
        __syncthreads();

        // ---- 4 WMMAs per barrier: 2 K-halves x 2 N-subtiles --------------
#pragma unroll
        for (int kk = 0; kk < 2; ++kk) {
            const v16bf af  = *(const v16bf*)AsF[kk][wm][lane];
            const v16bf bf0 = *(const v16bf*)BsF[kk][wn * 2 + 0][lane];
            const v16bf bf1 = *(const v16bf*)BsF[kk][wn * 2 + 1][lane];
            acc0 = __builtin_amdgcn_wmma_f32_16x16x32_bf16(false, af, false, bf0,
                                                           (short)0, acc0, false, false);
            acc1 = __builtin_amdgcn_wmma_f32_16x16x32_bf16(false, af, false, bf1,
                                                           (short)0, acc1, false, false);
        }
        __syncthreads();
    }

    // ---- epilogue + store (C layout: VGPR i -> row base+i, col lane&15) ----
    const int row0 = bm + wm * 16 + (lane >> 4) * 8;
#pragma unroll
    for (int s = 0; s < 2; ++s) {
        const int col = bn + (wn * 2 + s) * 16 + (lane & 15);
        if (col >= N) continue;
        const v8f acc = s ? acc1 : acc0;
#pragma unroll
        for (int i = 0; i < 8; ++i) {
            const int r = row0 + i;
            if (r < M) {
                float val = acc[i];
                if (rowdiv) val /= rowdiv[r];
                if (bias)   val += bias[col];
                if (Cacc)   val += Cacc[(size_t)r * ldc + col];
                if (relu)   val = fmaxf(val, 0.f);
                Cout[(size_t)r * ldc + col] = val;
            }
        }
    }
}

// ------------------------------ helpers ------------------------------------
__global__ void k_fill(float* p, float v, size_t n) {
    size_t i = (size_t)blockIdx.x * 256 + threadIdx.x;
    if (i < n) p[i] = v;
}

__global__ void k_build_A(const int* __restrict__ ei, float* A, int E) {
    int e = blockIdx.x * 256 + threadIdx.x;
    if (e < E) atomicAdd(&A[(size_t)ei[e] * LD + ei[E + e]], 1.f);
}

__global__ void k_setdiag(float* A, int n) {
    int i = blockIdx.x * 256 + threadIdx.x;
    if (i < n) A[(size_t)i * LD + i] = 1.f;
}

// deg[j] = max(#nonzero in column j, 1)
__global__ void k_degrees(const float* __restrict__ A, float* deg, int n) {
    int j = blockIdx.x * 256 + threadIdx.x;
    if (j >= n) return;
    int c = 0;
    for (int i = 0; i < n; ++i) c += (A[(size_t)i * LD + j] != 0.f);
    deg[j] = (float)(c > 0 ? c : 1);
}

// out[i,c] = max over j with A[j,i]!=0 of x[j,c]   (diag already set)
__global__ void k_colmax(const float* __restrict__ A, const float* __restrict__ x,
                         float* __restrict__ out, int n) {
    int i = blockIdx.x, c = threadIdx.x;           // HID threads
    float m = -1e30f;
    for (int j = 0; j < n; ++j) {
        float a = A[(size_t)j * LD + i];           // wave-uniform address
        if (a != 0.f) m = fmaxf(m, x[(size_t)j * HID + c]);
    }
    out[(size_t)i * HID + c] = m;
}

// out[i] = dot(X[i,:HID], w) + (*b)
__global__ void k_dotv(const float* __restrict__ X, const float* __restrict__ w,
                       const float* __restrict__ b, float* out, int n) {
    int i = blockIdx.x * 256 + threadIdx.x;
    if (i >= n) return;
    float s = b ? *b : 0.f;
    for (int c = 0; c < HID; ++c) s += X[(size_t)i * HID + c] * w[c];
    out[i] = s;
}

// Masked column softmax of leaky_relu(u[j]+vv[i]+attb) over rows j, mask A!=0.
__global__ __launch_bounds__(256) void k_att_softmax(
    const float* __restrict__ A, const float* __restrict__ u,
    const float* __restrict__ vv, const float* __restrict__ attb,
    float* __restrict__ S, int n)
{
    __shared__ float red[256];
    const int i = blockIdx.x, tid = threadIdx.x;
    const float vi = vv[i] + *attb;
    float mx = -1e30f;
    for (int j = tid; j < n; j += 256)
        if (A[(size_t)j * LD + i] != 0.f) {
            float t = u[j] + vi; t = t > 0.f ? t : 0.2f * t;
            mx = fmaxf(mx, t);
        }
    red[tid] = mx; __syncthreads();
    for (int s = 128; s > 0; s >>= 1) { if (tid < s) red[tid] = fmaxf(red[tid], red[tid + s]); __syncthreads(); }
    mx = red[0]; __syncthreads();
    float sm = 0.f;
    for (int j = tid; j < n; j += 256)
        if (A[(size_t)j * LD + i] != 0.f) {
            float t = u[j] + vi; t = t > 0.f ? t : 0.2f * t;
            sm += __expf(t - mx);
        }
    red[tid] = sm; __syncthreads();
    for (int s = 128; s > 0; s >>= 1) { if (tid < s) red[tid] += red[tid + s]; __syncthreads(); }
    const float inv = 1.f / red[0];
    for (int j = tid; j < n; j += 256) {
        float o = 0.f;
        if (A[(size_t)j * LD + i] != 0.f) {
            float t = u[j] + vi; t = t > 0.f ? t : 0.2f * t;
            o = __expf(t - mx) * inv;
        }
        S[(size_t)j * LD + i] = o;
    }
}

// fit[i] = sigmoid( sum_j mask[j,i]*a[j] - deg[i]*b[i] + c3[i] + le3b )
__global__ __launch_bounds__(256) void k_fitness(
    const float* __restrict__ A, const float* __restrict__ a,
    const float* __restrict__ b, const float* __restrict__ c3,
    const float* __restrict__ deg, const float* __restrict__ le3b,
    float* fit, int n)
{
    __shared__ float red[256];
    const int i = blockIdx.x, tid = threadIdx.x;
    float s = 0.f;
    for (int j = tid; j < n; j += 256)
        if (A[(size_t)j * LD + i] != 0.f) s += a[j];
    red[tid] = s; __syncthreads();
    for (int st = 128; st > 0; st >>= 1) { if (tid < st) red[tid] += red[tid + st]; __syncthreads(); }
    if (tid == 0) {
        float t = red[0] - deg[i] * b[i] + c3[i] + *le3b;
        fit[i] = 1.f / (1.f + __expf(-t));
    }
}

// Per-group top-k (stable, first-max-on-tie like lax.top_k). Tiny: ncur<=128.
__global__ void k_topk(const float* __restrict__ fit, int* perm, int ncur, int k) {
    const int g = blockIdx.x;
    if (threadIdx.x != 0) return;
    bool taken[256];
    for (int t = 0; t < ncur; ++t) taken[t] = false;
    for (int r = 0; r < k; ++r) {
        float best = -1e30f; int bi = 0;
        for (int t = 0; t < ncur; ++t)
            if (!taken[t]) {
                float f = fit[g * ncur + t];
                if (f > best) { best = f; bi = t; }
            }
        taken[bi] = true;
        perm[g * k + r] = g * ncur + bi;
    }
}

__global__ void k_gather_scale(const float* __restrict__ xnew,
                               const float* __restrict__ fit,
                               const int* __restrict__ perm,
                               float* out, int nout) {
    int idx = blockIdx.x * 256 + threadIdx.x;
    if (idx >= nout * HID) return;
    int r = idx / HID, c = idx % HID;
    int p = perm[r];
    out[idx] = xnew[(size_t)p * HID + c] * fit[p];
}

__global__ void k_gather_S(const float* __restrict__ S, const int* __restrict__ perm,
                           float* out, int n, int nout) {
    int r = blockIdx.x * 16 + threadIdx.x;   // out column
    int j = blockIdx.y * 16 + threadIdx.y;   // row
    if (r < nout && j < n) out[(size_t)j * LD + r] = S[(size_t)j * LD + perm[r]];
}

__global__ void k_gpool(const float* __restrict__ h, float* __restrict__ jbuf,
                        int npg, int off) {
    int g = blockIdx.x, c = threadIdx.x;     // HID threads
    float s = 0.f;
    for (int r = 0; r < npg; ++r) s += h[(size_t)(g * npg + r) * HID + c];
    jbuf[(size_t)g * (5 * HID) + off + c] = s / (float)npg;
}

// lin2 + log_softmax over 2 classes
__global__ void k_head(const float* __restrict__ z1, const float* __restrict__ w2,
                       const float* __restrict__ b2, float* out) {
    int g = threadIdx.x;
    if (g >= G) return;
    float z0 = b2[0], za = b2[1];
    for (int c = 0; c < HID; ++c) {
        float h = z1[(size_t)g * HID + c];
        z0 += h * w2[c]; za += h * w2[HID + c];
    }
    float m = fmaxf(z0, za);
    float lse = m + __logf(__expf(z0 - m) + __expf(za - m));
    out[g * 2 + 0] = z0 - lse;
    out[g * 2 + 1] = za - lse;
}

// ------------------------------ driver -------------------------------------
static void gemm(hipStream_t s,
                 const float* A, int lda, int tA,
                 const float* B, int ldb, int tB,
                 const float* Cacc, float* Cout, int ldc,
                 int M, int N, int K,
                 const float* rowdiv, const float* bias, int relu) {
    dim3 grid((N + BN - 1) / BN, (M + BM - 1) / BM);
    if (tA == 0 && tB == 0)
        k_gemm<0,0><<<grid, 256, 0, s>>>(A, lda, B, ldb, Cacc, Cout, ldc, M, N, K, rowdiv, bias, relu);
    else if (tA == 1 && tB == 0)
        k_gemm<1,0><<<grid, 256, 0, s>>>(A, lda, B, ldb, Cacc, Cout, ldc, M, N, K, rowdiv, bias, relu);
    else
        k_gemm<0,1><<<grid, 256, 0, s>>>(A, lda, B, ldb, Cacc, Cout, ldc, M, N, K, rowdiv, bias, relu);
}

extern "C" void kernel_launch(void* const* d_in, const int* in_sizes, int n_in,
                              void* d_out, int out_size, void* d_ws, size_t ws_size,
                              hipStream_t stream) {
    (void)n_in; (void)out_size; (void)ws_size;
    const float* x        = (const float*)d_in[0];
    const int*   ei       = (const int*)  d_in[1];
    const float* c0_wrel  = (const float*)d_in[2];
    const float* c0_brel  = (const float*)d_in[3];
    const float* c0_wroot = (const float*)d_in[4];
    const float* cw_rel   = (const float*)d_in[5];
    const float* cb_rel   = (const float*)d_in[6];
    const float* cw_root  = (const float*)d_in[7];
    const float* p_lin_w  = (const float*)d_in[8];
    const float* p_lin_b  = (const float*)d_in[9];
    const float* p_att_w  = (const float*)d_in[10];
    const float* p_att_b  = (const float*)d_in[11];
    const float* p_le1_w  = (const float*)d_in[12];
    const float* p_le1_b  = (const float*)d_in[13];
    const float* p_le2_w  = (const float*)d_in[14];
    const float* p_le3_w  = (const float*)d_in[15];
    const float* p_le3_b  = (const float*)d_in[16];
    const float* lin1_w   = (const float*)d_in[17];
    const float* lin1_b   = (const float*)d_in[18];
    const float* lin2_w   = (const float*)d_in[19];
    const float* lin2_b   = (const float*)d_in[20];

    float* ws = (float*)d_ws;
    const size_t BIG = (size_t)LD * LD;
    float* bufA  = ws;             // adjacency (A, then A2 in place)
    float* bufS  = ws + 1 * BIG;   // attention S
    float* bufSS = ws + 2 * BIG;   // S_sel
    float* bufT  = ws + 3 * BIG;   // T1 = A @ S_sel
    float* p     = ws + 4 * BIG;
    const size_t NH = (size_t)N0 * HID;
    float* h0 = p;   p += NH;
    float* h1 = p;   p += NH;
    float* tmpH = p; p += NH;
    float* aggB = p; p += NH;
    float* xnew = p; p += NH;
    float* cmax = p; p += NH;
    float* qB   = p; p += NH;
    float* deg  = p; p += N0;
    float* uB   = p; p += N0;
    float* vB   = p; p += N0;
    float* aV   = p; p += N0;
    float* bV   = p; p += N0;
    float* cV   = p; p += N0;
    float* fit  = p; p += N0;
    float* jbuf = p; p += (size_t)G * 5 * HID;
    float* z1   = p; p += (size_t)G * HID;
    int*   perm = (int*)p;

    const int E = in_sizes[1] / 2;

    // adjacency from edge list
    k_fill<<<(int)((BIG + 255) / 256), 256, 0, stream>>>(bufA, 0.f, BIG);
    k_build_A<<<(E + 255) / 256, 256, 0, stream>>>(ei, bufA, E);

    int n = N0, ncur = N0 / G, pi = 0;
    float* hc = h0; float* hn = h1;

    // conv0: h = relu((A^T x / deg) W_rel^T + b + x W_root^T)
    k_degrees<<<(n + 255) / 256, 256, 0, stream>>>(bufA, deg, n);
    gemm(stream, bufA, LD, 1, x, 64, 0, nullptr, aggB, 64, n, 64, n, deg, nullptr, 0);
    gemm(stream, aggB, 64, 0, c0_wrel, 64, 1, nullptr, tmpH, HID, n, HID, 64, nullptr, c0_brel, 0);
    gemm(stream, x, 64, 0, c0_wroot, 64, 1, tmpH, hc, HID, n, HID, 64, nullptr, nullptr, 1);
    k_gpool<<<G, HID, 0, stream>>>(hc, jbuf, ncur, 0);

    const int KS[2] = {103, 83};
    for (int i = 0; i < 4; ++i) {
        k_degrees<<<(n + 255) / 256, 256, 0, stream>>>(bufA, deg, n);
        gemm(stream, bufA, LD, 1, hc, HID, 0, nullptr, aggB, HID, n, HID, n, deg, nullptr, 0);
        gemm(stream, aggB, HID, 0, cw_rel + (size_t)i * HID * HID, HID, 1,
             nullptr, tmpH, HID, n, HID, HID, nullptr, cb_rel + (size_t)i * HID, 0);
        gemm(stream, hc, HID, 0, cw_root + (size_t)i * HID * HID, HID, 1,
             tmpH, hn, HID, n, HID, HID, nullptr, nullptr, 1);
        { float* t = hc; hc = hn; hn = t; }
        k_gpool<<<G, HID, 0, stream>>>(hc, jbuf, ncur, (i + 1) * HID);

        if (i % 2 == 0 && i < 3) {                     // ASAP pool
            const int k = KS[pi];
            k_setdiag<<<(n + 255) / 256, 256, 0, stream>>>(bufA, n);
            k_degrees<<<(n + 255) / 256, 256, 0, stream>>>(bufA, deg, n);
            k_colmax<<<n, HID, 0, stream>>>(bufA, hc, cmax, n);
            gemm(stream, cmax, HID, 0, p_lin_w + (size_t)pi * HID * HID, HID, 1,
                 nullptr, qB, HID, n, HID, HID, nullptr, p_lin_b + (size_t)pi * HID, 0);
            k_dotv<<<(n + 255) / 256, 256, 0, stream>>>(hc, p_att_w + (size_t)pi * 2 * HID + HID, nullptr, uB, n);
            k_dotv<<<(n + 255) / 256, 256, 0, stream>>>(qB, p_att_w + (size_t)pi * 2 * HID, nullptr, vB, n);
            k_att_softmax<<<n, 256, 0, stream>>>(bufA, uB, vB, p_att_b + pi, bufS, n);
            gemm(stream, bufS, LD, 1, hc, HID, 0, nullptr, xnew, HID, n, HID, n, nullptr, nullptr, 0);
            k_dotv<<<(n + 255) / 256, 256, 0, stream>>>(xnew, p_le1_w + (size_t)pi * HID, p_le1_b + pi, aV, n);
            k_dotv<<<(n + 255) / 256, 256, 0, stream>>>(xnew, p_le2_w + (size_t)pi * HID, nullptr, bV, n);
            k_dotv<<<(n + 255) / 256, 256, 0, stream>>>(xnew, p_le3_w + (size_t)pi * HID, nullptr, cV, n);
            k_fitness<<<n, 256, 0, stream>>>(bufA, aV, bV, cV, deg, p_le3_b + pi, fit, n);
            k_topk<<<G, 32, 0, stream>>>(fit, perm, ncur, k);
            const int nout = G * k;
            k_gather_scale<<<(nout * HID + 255) / 256, 256, 0, stream>>>(xnew, fit, perm, hn, nout);
            { float* t = hc; hc = hn; hn = t; }
            k_gather_S<<<dim3((nout + 15) / 16, (n + 15) / 16), dim3(16, 16), 0, stream>>>(bufS, perm, bufSS, n, nout);
            // T1 = A @ S_sel  (compute-bound: run on WMMA)
            gemm(stream, bufA, LD, 0, bufSS, LD, 0, nullptr, bufT, LD, n, nout, n, nullptr, nullptr, 0);
            // A2 = S_sel^T @ T1, overwrite A in place (A no longer needed)
            gemm(stream, bufSS, LD, 1, bufT, LD, 0, nullptr, bufA, LD, nout, nout, n, nullptr, nullptr, 0);
            k_setdiag<<<(nout + 255) / 256, 256, 0, stream>>>(bufA, nout);
            n = nout; ncur = k; ++pi;
        }
    }

    // head: z1 = relu(j @ lin1_w^T + b1); out = log_softmax(z1 @ lin2_w^T + b2)
    gemm(stream, jbuf, 5 * HID, 0, lin1_w, 5 * HID, 1, nullptr, z1, HID,
         G, HID, 5 * HID, nullptr, lin1_b, 1);
    k_head<<<1, 32, 0, stream>>>(z1, lin2_w, lin2_b, (float*)d_out);
}
